// BotRGCN_53300544144088
// MI455X (gfx1250) — matrix operand order repaired and checked by reading, not compiled, not run on hardware
//
#include <hip/hip_runtime.h>

// ===========================================================================
// BotRGCN forward for MI455X (gfx1250, wave32, WMMA).
//  - Dense linears via v_wmma_f32_16x16x32_f16 (fp32 accum).
//    * Weights are pre-packed per GEMM into the exact WMMA B-fragment layout
//      (f16, zero-padded in K and N) -> inner loop B fetch = one contiguous
//      32B v16h load per lane (2x global_load_b128, coalesced).
//    * All A operands have row strides padded to multiples of 32 floats and
//      guaranteed-zero pad columns -> A fetch = 4x b128 + cvt_pk, no guards,
//      no ragged-K tail. Inner loop is branch-free: loads + cvt + wmma.
//  - GraphConv aggregation: L2-resident atomic scatter-add (agg <= 51MB <<
//    192MB L2), then GEMM-accumulate agg @ W_rel into a shared neigh buffer.
//  - The 8 (resp 3) per-layer root terms are pre-summed into one weight/bias.
// ===========================================================================

typedef __attribute__((ext_vector_type(16))) _Float16 v16h;
typedef __attribute__((ext_vector_type(8)))  float    v8f;

// ---------------------------------------------------------------------------
__global__ void __launch_bounds__(256) k_zero(float* __restrict__ p, long long n) {
  long long i = (long long)blockIdx.x * blockDim.x + threadIdx.x;
  long long s = (long long)gridDim.x * blockDim.x;
  for (; i < n; i += s) p[i] = 0.f;
}

__global__ void __launch_bounds__(256) k_copy(float* __restrict__ o, const float* __restrict__ a, int n) {
  int i = blockIdx.x * blockDim.x + threadIdx.x;
  if (i < n) o[i] = a[i];
}

__global__ void __launch_bounds__(256) k_add(float* __restrict__ o, const float* __restrict__ a, int n) {
  int i = blockIdx.x * blockDim.x + threadIdx.x;
  if (i < n) o[i] += a[i];
}

// dst[row, 0:ldd] = [src[row, 0:D] | zeros]   (zero-padded staging copy)
__global__ void __launch_bounds__(256) k_padcopy(const float* __restrict__ src, int D,
                                                 float* __restrict__ dst, int ldd,
                                                 long long M) {
  long long total = M * ldd;
  long long s = (long long)gridDim.x * blockDim.x;
  for (long long i = (long long)blockIdx.x * blockDim.x + threadIdx.x; i < total; i += s) {
    long long row = i / ldd;
    int col = (int)(i - row * ldd);
    dst[i] = (col < D) ? src[row * D + col] : 0.f;
  }
}

// Pack W (K x N fp32 row-major) into WMMA B fragments, zero-padded to
// (ktiles*32) x (ntiles*16).  Fragment layout (contiguous k order, as k_gemm loads A):
//   lane l: column n = ntile*16 + (l&15); half jj -> K = kt*32 + (l>>4)*16 + jj
//   P[(((ntile*ktiles)+kt)*32 + lane)*16 + jj]
__global__ void __launch_bounds__(256) k_packB(const float* __restrict__ W,
                                               _Float16* __restrict__ P,
                                               int K, int N, int ntiles, int ktiles) {
  int total = ktiles * ntiles * 512;
  int s = gridDim.x * blockDim.x;
  for (int i = blockIdx.x * blockDim.x + threadIdx.x; i < total; i += s) {
    int jj    = i & 15;
    int lane  = (i >> 4) & 31;
    int tile  = i >> 9;              // ntile*ktiles + kt
    int kt    = tile % ktiles;
    int ntile = tile / ktiles;
    int n  = ntile * 16 + (lane & 15);
    int kg = kt * 32 + (lane >> 4) * 16 + jj;
    float v = (n < N && kg < K) ? W[(size_t)kg * N + n] : 0.f;
    P[i] = (_Float16)v;
  }
}

// agg[dst[e], f] += ew[e] * X[src[e], f]   (f < D; padded row strides)
__global__ void __launch_bounds__(256) k_scatter(const int* __restrict__ ei, int E,
                                                 const float* __restrict__ ew,
                                                 const float* __restrict__ X, int ldx,
                                                 int D,
                                                 float* __restrict__ AGG, int lda) {
  const int* src = ei;
  const int* dst = ei + E;
  long long total = (long long)E * D;
  long long s = (long long)gridDim.x * blockDim.x;
  for (long long i = (long long)blockIdx.x * blockDim.x + threadIdx.x; i < total; i += s) {
    int e = (int)(i / D);
    int f = (int)(i - (long long)e * D);
    atomicAdd(&AGG[(long long)dst[e] * lda + f], ew[e] * X[(long long)src[e] * ldx + f]);
  }
}

// Row-wise L2 normalize over cols [0,D) with eps=1e-12 (one wave32 per row).
__global__ void __launch_bounds__(32) k_rownorm(float* __restrict__ X, int D, int ld) {
  float* R = X + (long long)blockIdx.x * ld;
  float ss = 0.f;
  for (int f = threadIdx.x; f < D; f += 32) { float v = R[f]; ss += v * v; }
#pragma unroll
  for (int o = 16; o > 0; o >>= 1) ss += __shfl_xor(ss, o, 32);
  float inv = 1.f / fmaxf(sqrtf(ss), 1e-12f);
  for (int f = threadIdx.x; f < D; f += 32) R[f] *= inv;
}

// ---------------------------------------------------------------------------
// C[:, colOff:colOff+N] (ldc) (+)= act( A(M x ktiles*32, lda) @ Bpack + bias )
// A: row stride lda (multiple of 32), zero-padded beyond the logical K.
// Bpack: pre-packed f16 fragments (see k_packB).
// One wave per 16x16 tile; M is always a multiple of 16 here.
__global__ void __launch_bounds__(32)
k_gemm(const float* __restrict__ A, int lda,
       const _Float16* __restrict__ Bp, int ktiles,
       const float* __restrict__ bias,        // may be null
       float* __restrict__ C, int ldc, int colOff,
       int M, int N, int acc, int act)
{
  const int lane  = threadIdx.x;
  const int mt    = blockIdx.y * 16;
  const int ntile = blockIdx.x;
  const int nt    = ntile * 16;
  const int mrow  = lane & 15;
  const int half  = lane >> 4;       // 0 or 1
  const int ncol  = lane & 15;
  const int mbase = half << 3;

  const int  nw     = nt + ncol;
  const bool nvalid = (nw < N);
  const int  nwc    = nvalid ? nw : (N - 1);

  v8f c;
  if (acc) {
#pragma unroll
    for (int j = 0; j < 8; ++j) {
      float v = C[(size_t)(mt + mbase + j) * ldc + colOff + nwc];
      c[j] = nvalid ? v : 0.f;
    }
  } else {
#pragma unroll
    for (int j = 0; j < 8; ++j) c[j] = 0.f;
  }

  const float*    Arow = A + (size_t)(mt + mrow) * lda + half * 16;
  const _Float16* Bt   = Bp + ((size_t)ntile * ktiles) * 512 + lane * 16;

#pragma unroll 2
  for (int kt = 0; kt < ktiles; ++kt) {
    v16h b = *(const v16h*)(Bt + (size_t)kt * 512);   // 32B contiguous per lane
    const float* Ak = Arow + kt * 32;
    v16h a;
#pragma unroll
    for (int j = 0; j < 4; ++j) {
      float2 p = *(const float2*)(Ak + 2 * j);
      a[2 * j]     = (_Float16)p.x;
      a[2 * j + 1] = (_Float16)p.y;
    }
#pragma unroll
    for (int j = 0; j < 4; ++j) {
      float2 p = *(const float2*)(Ak + 8 + 2 * j);
      a[8 + 2 * j]     = (_Float16)p.x;
      a[8 + 2 * j + 1] = (_Float16)p.y;
    }
    c = __builtin_amdgcn_wmma_f32_16x16x32_f16(false, a, false, b, (short)0, c, false, false);
  }

  if (nvalid) {  // single uniform-per-lane guard; stores exec-masked
    float bv = bias ? bias[nw] : 0.f;
#pragma unroll
    for (int j = 0; j < 8; ++j) {
      float v = c[j] + bv;
      if (act) v = (v >= 0.f) ? v : 0.01f * v;
      C[(size_t)(mt + mbase + j) * ldc + colOff + nw] = v;
    }
  }
}

// ===========================================================================
// Host orchestration
// ===========================================================================

// Param leaves in canonical = sorted-key (jax tree_flatten) order.
static const int canonSize[102] = {
  23040,30, 3600,60, 150,30, 240,2, 8100,90, 420,30, 46080,60, 280,40, 30720,40,
  14400,120, 440,40, 14400,120,
  1800,30,
  7200,60,3600, 7200,60,3600, 7200,60,3600,
  1800,30,
  4050,45,
  10800,90,8100, 10800,90,8100, 10800,90,8100,
  4050,45,
  7200,60,
  14400,120,14400, 14400,120,14400,
  10800,120,14400, 10800,120,14400, 10800,120,14400,
  7200,120,14400, 7200,120,14400, 7200,120,14400,
  14400,120,14400, 14400,120,14400,
  10800,120,14400, 10800,120,14400, 10800,120,14400,
  7200,120,14400, 7200,120,14400, 7200,120,14400,
  7200,60
};

// Insertion-order (raw dict order in _init_params) -> canonical index.
static const short insCanon[102] = {
  16,17, 20,21, 14,15, 18,19,
  10,11, 12,13, 8,9,
  4,5, 0,1, 2,3,
  52,53,54, 55,56,57, 58,59,60, 61,62,63, 64,65,66, 67,68,69, 70,71,72, 73,74,75,
  76,77,78, 79,80,81, 82,83,84, 85,86,87, 88,89,90, 91,92,93, 94,95,96, 97,98,99,
  39,40,41, 26,27,28, 42,43,44, 29,30,31, 45,46,47, 32,33,34,
  100,101, 50,51, 48,49, 37,38, 35,36, 24,25,
  22,23, 6,7
};

extern "C" void kernel_launch(void* const* d_in, const int* in_sizes, int n_in,
                              void* d_out, int out_size, void* d_ws, size_t ws_size,
                              hipStream_t stream)
{
  (void)out_size; (void)ws_size;
  const int NU = 100000, NT = 100000, NL = 10000;
  const int UP = 128, TP = 96, LP = 64;   // padded strides for 120 / 90 / 60

  const float* user_desc  = (const float*)d_in[0];
  const float* user_num   = (const float*)d_in[1];
  const float* user_cat   = (const float*)d_in[2];
  const float* tweet_num  = (const float*)d_in[3];
  const float* tweet_text = (const float*)d_in[4];
  const float* list_num   = (const float*)d_in[5];
  const float* list_desc  = (const float*)d_in[6];
  // edges: ei at 7,9,...,33 ; ew at 8,10,...,34 (setup_inputs order)

  // ---- resolve param leaf pointers -> canonical table -------------------
  const float* par[102];
  if (n_in >= 35 + 102) {
    bool sortedOK = true;
    for (int j = 0; j < 102; ++j)
      if (in_sizes[35 + j] != canonSize[j]) { sortedOK = false; break; }
    if (sortedOK) {
      for (int j = 0; j < 102; ++j) par[j] = (const float*)d_in[35 + j];
    } else {
      for (int p = 0; p < 102; ++p) par[insCanon[p]] = (const float*)d_in[35 + p];
    }
  } else {
    const float* base = (const float*)d_in[n_in - 1];
    size_t off = 0;
    for (int j = 0; j < 102; ++j) { par[j] = base + off; off += (size_t)canonSize[j]; }
  }

  // ---- workspace layout (floats; every block a multiple of 8 for 32B align)
  float* w0 = (float*)d_ws;
  float* w = w0;
  auto take = [&](size_t n) { float* p = w; w += n; return p; };
  float* padbuf    = take((size_t)NU * 32);   // staged num/cat inputs (K pad 32)
  float* ucat      = take((size_t)NU * UP);   // reused later as out120
  float* user_x    = take((size_t)NU * UP);   // reused later as ux2
  float* tcat      = take((size_t)NT * TP);
  float* tweet_x   = take((size_t)NT * TP);
  float* lcat      = take((size_t)NL * LP);
  float* list_x    = take((size_t)NL * LP);
  float* agg       = take((size_t)NU * UP);
  float* neigh     = take((size_t)NU * UP);
  float* new_user  = take((size_t)NU * UP);
  float* new_tweet = take((size_t)NT * TP);
  float* new_list  = take((size_t)NL * LP);
  float* wsum      = take(14400);
  float* bsum      = take(128);
  _Float16* bpack  = (_Float16*)take(65536);  // 131072 halves, max need 49152

  // ---- launch helpers ---------------------------------------------------
  auto zero  = [&](float* p, long long n) { k_zero<<<2048, 256, 0, stream>>>(p, n); };
  auto copyv = [&](float* o, const float* a, int n) { k_copy<<<(n + 255) / 256, 256, 0, stream>>>(o, a, n); };
  auto addv  = [&](float* o, const float* a, int n) { k_add<<<(n + 255) / 256, 256, 0, stream>>>(o, a, n); };
  auto gemm = [&](const float* A, int lda, const float* W, const float* bias,
                  float* C, int ldc, int colOff, int M, int N, int K, int acc, int act) {
    int ktiles = (K + 31) / 32;
    int ntiles = (N + 15) / 16;
    int total  = ktiles * ntiles * 512;
    k_packB<<<(total + 255) / 256, 256, 0, stream>>>(W, bpack, K, N, ntiles, ktiles);
    dim3 g((unsigned)ntiles, (unsigned)(M / 16));
    k_gemm<<<g, 32, 0, stream>>>(A, lda, bpack, ktiles, bias, C, ldc, colOff, M, N, acc, act);
  };
  auto padcopy = [&](const float* src, int D, float* dst, int ldd, long long M) {
    k_padcopy<<<2048, 256, 0, stream>>>(src, D, dst, ldd, M);
  };
  auto scatter = [&](int eidx, const float* X, int ldx, int D, float* AGG, int lda, int rowsDst) {
    const int* ei = (const int*)d_in[eidx];
    int E = in_sizes[eidx] / 2;
    const float* ew = (const float*)d_in[eidx + 1];
    zero(AGG, (long long)rowsDst * lda);
    k_scatter<<<4096, 256, 0, stream>>>(ei, E, ew, X, ldx, D, AGG, lda);
  };
  auto rownorm = [&](float* X, int M, int D, int ld) { k_rownorm<<<M, 32, 0, stream>>>(X, D, ld); };

  // Zero the whole workspace once: establishes the all-pads-zero invariant.
  zero(w0, (long long)(w - w0));

  // ======================= encoders ======================================
  gemm(user_desc, 768, par[16], par[17], ucat, UP,  0, NU, 40, 768, 0, 1);
  padcopy(user_num, 11, padbuf, 32, NU);
  gemm(padbuf, 32, par[20], par[21], ucat, UP, 40, NU, 40, 11, 0, 1);
  padcopy(user_cat, 7, padbuf, 32, NU);
  gemm(padbuf, 32, par[14], par[15], ucat, UP, 80, NU, 40, 7, 0, 1);
  gemm(ucat, UP, par[18], par[19], user_x, UP, 0, NU, 120, 120, 0, 1);
  rownorm(user_x, NU, 120, UP);

  padcopy(tweet_num, 14, padbuf, 32, NT);
  gemm(padbuf, 32, par[10], par[11], tcat, TP, 0, NT, 30, 14, 0, 1);
  gemm(tweet_text, 768, par[12], par[13], tcat, TP, 30, NT, 60, 768, 0, 1);
  gemm(tcat, TP, par[8], par[9], tweet_x, TP, 0, NT, 90, 90, 0, 1);
  rownorm(tweet_x, NT, 90, TP);

  padcopy(list_num, 5, padbuf, 32, NL);
  gemm(padbuf, 32, par[4], par[5], lcat, LP, 0, NL, 30, 5, 0, 1);
  gemm(list_desc, 768, par[0], par[1], lcat, LP, 30, NL, 30, 768, 0, 1);
  gemm(lcat, LP, par[2], par[3], list_x, LP, 0, NL, 60, 60, 0, 1);
  rownorm(list_x, NL, 60, LP);

  // ======================= user layer (shared L1/L2) =====================
  const int uEdge[8] = {7, 9, 11, 13, 15, 17, 19, 21};
  auto user_layer = [&](int sageBase, const float* ux, const float* tx, const float* lx,
                        float* outbuf) {
    copyv(wsum, par[sageBase + 2], 14400);
    for (int k = 1; k < 8; ++k) addv(wsum, par[sageBase + 3 * k + 2], 14400);
    copyv(bsum, par[sageBase + 1], 120);
    for (int k = 1; k < 8; ++k) addv(bsum, par[sageBase + 3 * k + 1], 120);
    zero(neigh, (long long)NU * UP);   // re-establish pad invariant for stride UP
    gemm(ux, UP, wsum, bsum, neigh, UP, 0, NU, 120, 120, 0, 0);
    for (int k = 0; k < 8; ++k) {
      int D = (k < 2) ? 120 : (k < 5) ? 90 : 60;
      int Dp = (k < 2) ? UP : (k < 5) ? TP : LP;
      const float* Xs = (k < 2) ? ux : (k < 5) ? tx : lx;
      int ldx = (k < 2) ? UP : (k < 5) ? TP : LP;
      scatter(uEdge[k], Xs, ldx, D, agg, Dp, NU);
      gemm(agg, Dp, par[sageBase + 3 * k], nullptr, neigh, UP, 0, NU, 120, D, 1, 0);
    }
    gemm(ux,    UP, par[100], par[101], outbuf, UP,  0, NU, 60, 120, 0, 1);
    gemm(neigh, UP, par[50],  par[51],  outbuf, UP, 60, NU, 60, 120, 0, 1);
    rownorm(outbuf, NU, 120, UP);
  };

  // ======================= layer 1 =======================================
  user_layer(52, user_x, tweet_x, list_x, new_user);

  // tweet layer 1
  copyv(wsum, par[41], 8100); addv(wsum, par[44], 8100); addv(wsum, par[47], 8100);
  copyv(bsum, par[40], 90);   addv(bsum, par[43], 90);   addv(bsum, par[46], 90);
  zero(neigh, (long long)NT * TP);
  gemm(tweet_x, TP, wsum, bsum, neigh, TP, 0, NT, 90, 90, 0, 0);
  {
    const int tE[3] = {23, 25, 27}; const int tR[3] = {39, 42, 45};
    for (int i = 0; i < 3; ++i) {
      scatter(tE[i], user_x, UP, 120, agg, UP, NT);
      gemm(agg, UP, par[tR[i]], nullptr, neigh, TP, 0, NT, 90, 120, 1, 0);
    }
  }
  gemm(tweet_x, TP, par[48], par[49], new_tweet, TP,  0, NT, 45, 90, 0, 1);
  gemm(neigh,   TP, par[37], par[38], new_tweet, TP, 45, NT, 45, 90, 0, 1);
  rownorm(new_tweet, NT, 90, TP);

  // list layer 1
  copyv(wsum, par[28], 3600); addv(wsum, par[31], 3600); addv(wsum, par[34], 3600);
  copyv(bsum, par[27], 60);   addv(bsum, par[30], 60);   addv(bsum, par[33], 60);
  zero(neigh, (long long)NL * LP);
  gemm(list_x, LP, wsum, bsum, neigh, LP, 0, NL, 60, 60, 0, 0);
  {
    const int lE[3] = {29, 31, 33}; const int lR[3] = {26, 29, 32};
    for (int i = 0; i < 3; ++i) {
      scatter(lE[i], user_x, UP, 120, agg, UP, NL);
      gemm(agg, UP, par[lR[i]], nullptr, neigh, LP, 0, NL, 60, 120, 1, 0);
    }
  }
  gemm(list_x, LP, par[35], par[36], new_list, LP,  0, NL, 30, 60, 0, 1);
  gemm(neigh,  LP, par[24], par[25], new_list, LP, 30, NL, 30, 60, 0, 1);
  rownorm(new_list, NL, 60, LP);

  // ======================= layer 2 (user only) ===========================
  float* ux2 = user_x;   // user_x no longer needed; same stride UP
  user_layer(76, new_user, new_tweet, new_list, ux2);

  // ======================= head ==========================================
  float* out120 = ucat;  // ucat no longer needed; same stride UP, pads still 0
  gemm(ux2,    UP, par[22], par[23], out120,        UP, 0, NU, 120, 120, 0, 1);
  gemm(out120, UP, par[6],  par[7],  (float*)d_out,  2, 0, NU,   2, 120, 0, 0);
}
